// BaseModel_10677288698164
// MI455X (gfx1250) — matrix operand order repaired and checked by reading, not compiled, not run on hardware
//
#include <hip/hip_runtime.h>

// ---------------------------------------------------------------------------
// GRU pose rollout for MI455X (gfx1250, wave32, WMMA bf16 -> f32 accumulate)
// B=2048 T=144 D=135 H=1024 J=15 JD=9 SEED_LEN=120 PRED_FRAMES=24
// v2: paired N-tiles share one A fragment (halves LDS traffic per WMMA),
//     global_prefetch on the weight stream.
// ---------------------------------------------------------------------------

typedef __attribute__((ext_vector_type(16))) __bf16 v16bf;
typedef __attribute__((ext_vector_type(8)))  __bf16 v8bf;
typedef __attribute__((ext_vector_type(8)))  float  v8f;

__device__ __forceinline__ unsigned short f2bf(float f) {
    unsigned u = __float_as_uint(f);
    unsigned r = ((u >> 16) & 1u) + 0x7FFFu;   // round-to-nearest-even
    return (unsigned short)((u + r) >> 16);
}
__device__ __forceinline__ float bf2f(unsigned short h) {
    return __uint_as_float(((unsigned)h) << 16);
}

union AFrag { v16bf v; v8bf h[2]; };

// A-matrix 16x32 bf16 fragment from LDS (row-major, stride in halfs).
// ISA 7.12.2: lanes 0-15 (hi=0): M=lane, K = {k0..k0+7, k0+16..k0+23};
// lanes 16-31 (hi=1): K = {k0+8..k0+15, k0+24..k0+31}.
__device__ __forceinline__ v16bf load_a(const unsigned short* base, int stride,
                                        int m, int hi, int k0) {
    AFrag f;
    const unsigned short* p = base + m * stride + k0 + hi * 8;
    f.h[0] = *(const v8bf*)(p);        // ds_load_b128
    f.h[1] = *(const v8bf*)(p + 16);   // ds_load_b128
    return f.v;
}

// B-matrix 32x16 fragment, pre-swizzled in workspace: [tile][lane][16 halfs]
// -> one coalesced 32B load per lane.
__device__ __forceinline__ v16bf load_b(const unsigned short* pk, long tile, int lane) {
    return *(const v16bf*)(pk + (tile * 32 + (long)lane) * 16);
}

__device__ __forceinline__ v8f wmma_bf16(v16bf a, v16bf b, v8f c) {
    return __builtin_amdgcn_wmma_f32_16x16x32_bf16(
        /*neg_a=*/false, a, /*neg_b=*/false, b,
        /*c_mod=*/(short)0, c, /*reuse_a=*/false, /*reuse_b=*/false);
}

__device__ __forceinline__ void prefetch_b(const unsigned short* pk, long tile, int lane) {
    __builtin_prefetch((const void*)(pk + (tile * 32 + (long)lane) * 16), 0, 1);
}

// ---------------------------------------------------------------------------
// Prep kernel: pack a weight matrix (viewed as B[k][n] = src[n*sn + k*sk]) into
// bf16 B-fragment tiles, zero-padding K and N.
// ---------------------------------------------------------------------------
__global__ void pack_b_kernel(const float* __restrict__ src,
                              unsigned short* __restrict__ dst,
                              int NT, int KT, int N_src, int K_src,
                              long sn, long sk) {
    long idx = (long)blockIdx.x * blockDim.x + threadIdx.x;
    long total = (long)NT * KT * 512;          // 32 lanes * 16 halfs per tile
    if (idx >= total) return;
    int  s  = (int)(idx & 15);
    int  l  = (int)((idx >> 4) & 31);
    long t  = idx >> 9;
    int  kt = (int)(t % KT);
    int  nt = (int)(t / KT);
    int  n  = nt * 16 + (l & 15);
    int  k  = kt * 32 + ((s >> 3) << 4) + ((l >> 4) << 3) + (s & 7);
    float v = (n < N_src && k < K_src) ? src[(long)n * sn + (long)k * sk] : 0.0f;
    dst[idx] = f2bf(v);
}

// ---------------------------------------------------------------------------
// Persistent rollout: block = 256 threads (8 wave32s) owns 16 batch rows for
// all 24 frames.  State in LDS; bf16 weights stream from L2.
// ---------------------------------------------------------------------------
__global__ __launch_bounds__(256) void gru_rollout(
    const float* __restrict__ poses,
    const float* __restrict__ b_ih, const float* __restrict__ b_hh,
    const float* __restrict__ bp,   const float* __restrict__ b1,
    const float* __restrict__ b2,
    const unsigned short* __restrict__ pkWih,
    const unsigned short* __restrict__ pkWhh,
    const unsigned short* __restrict__ pkWp,
    const unsigned short* __restrict__ pkW1,
    const unsigned short* __restrict__ pkW2,
    float* __restrict__ out) {
    extern __shared__ __align__(16) unsigned short smem[];
    unsigned short* h0  = smem;                     // 16*1024 bf16 (ping)
    unsigned short* h1  = h0  + 16 * 1024;          // 16*1024 bf16 (pong)
    unsigned short* hid = h1  + 16 * 1024;          // 16*1024 bf16
    unsigned short* ub  = hid + 16 * 1024;          // 8 waves * 16*128 bf16
    float*          xf  = (float*)(ub + 8 * 16 * 128);       // 16*160 f32
    unsigned short* xb  = (unsigned short*)(xf + 16 * 160);  // 16*160 bf16

    const int tid  = threadIdx.x;
    const int w    = tid >> 5;
    const int lane = tid & 31;
    const int hi   = lane >> 4;
    const int lm   = lane & 15;
    const int b0   = blockIdx.x * 16;

    // ---- init: h = 0, x = poses[:, SEED_LEN-1] (padded 135 -> 160) ----
    for (int i = tid; i < 16 * 1024; i += 256) h0[i] = 0;
    for (int i = tid; i < 16 * 160; i += 256) {
        int m = i / 160, d = i % 160;
        float v = (d < 135) ? poses[((long)(b0 + m) * 144 + 119) * 135 + d] : 0.0f;
        xf[i] = v;
        xb[i] = f2bf(v);
    }
    __syncthreads();

    for (int f = 0; f < 24; ++f) {
        unsigned short* cur = (f & 1) ? h1 : h0;
        unsigned short* nxt = (f & 1) ? h0 : h1;

        // ============ Phase A: GRU gates, tiles processed in pairs ============
        for (int pp = 0; pp < 4; ++pp) {
            int ta = w * 8 + pp * 2;
            int tb = ta + 1;
            int ca = ta * 16 + lm;
            int cb = tb * 16 + lm;
            float br0  = b_ih[ca] + b_hh[ca],               br1  = b_ih[cb] + b_hh[cb];
            float bz0  = b_ih[1024 + ca] + b_hh[1024 + ca], bz1  = b_ih[1024 + cb] + b_hh[1024 + cb];
            float bin0 = b_ih[2048 + ca],                   bin1 = b_ih[2048 + cb];
            float bhn0 = b_hh[2048 + ca],                   bhn1 = b_hh[2048 + cb];
            v8f ar0, az0, ain0, ahn0, ar1, az1, ain1, ahn1;
            for (int i = 0; i < 8; ++i) {
                ar0[i] = br0;  az0[i] = bz0;  ain0[i] = bin0; ahn0[i] = bhn0;
                ar1[i] = br1;  az1[i] = bz1;  ain1[i] = bin1; ahn1[i] = bhn1;
            }
            // x contribution: K = 160 (135 zero-padded), one A serves 6 WMMAs
            for (int kt = 0; kt < 5; ++kt) {
                v16bf a = load_a(xb, 160, lm, hi, kt * 32);
                ar0  = wmma_bf16(a, load_b(pkWih, (long)(ta)       * 5 + kt, lane), ar0);
                ar1  = wmma_bf16(a, load_b(pkWih, (long)(tb)       * 5 + kt, lane), ar1);
                az0  = wmma_bf16(a, load_b(pkWih, (long)(ta + 64)  * 5 + kt, lane), az0);
                az1  = wmma_bf16(a, load_b(pkWih, (long)(tb + 64)  * 5 + kt, lane), az1);
                ain0 = wmma_bf16(a, load_b(pkWih, (long)(ta + 128) * 5 + kt, lane), ain0);
                ain1 = wmma_bf16(a, load_b(pkWih, (long)(tb + 128) * 5 + kt, lane), ain1);
            }
            // h contribution: K = 1024
            for (int kt = 0; kt < 32; ++kt) {
                // keep the L2-resident weight stream warm a few k-steps ahead
                prefetch_b(pkWhh, (long)(ta) * 32 + kt + 8, lane);
                prefetch_b(pkWhh, (long)(ta + 64) * 32 + kt + 8, lane);
                prefetch_b(pkWhh, (long)(ta + 128) * 32 + kt + 8, lane);
                v16bf a = load_a(cur, 1024, lm, hi, kt * 32);
                ar0  = wmma_bf16(a, load_b(pkWhh, (long)(ta)       * 32 + kt, lane), ar0);
                ar1  = wmma_bf16(a, load_b(pkWhh, (long)(tb)       * 32 + kt, lane), ar1);
                az0  = wmma_bf16(a, load_b(pkWhh, (long)(ta + 64)  * 32 + kt, lane), az0);
                az1  = wmma_bf16(a, load_b(pkWhh, (long)(tb + 64)  * 32 + kt, lane), az1);
                ahn0 = wmma_bf16(a, load_b(pkWhh, (long)(ta + 128) * 32 + kt, lane), ahn0);
                ahn1 = wmma_bf16(a, load_b(pkWhh, (long)(tb + 128) * 32 + kt, lane), ahn1);
            }
            // gate math on the 16x16 f32 accumulator layout (VGPR i -> row hi*8+i)
            for (int i = 0; i < 8; ++i) {
                int   m  = hi * 8 + i;
                float h0v = bf2f(cur[m * 1024 + ca]);
                float h1v = bf2f(cur[m * 1024 + cb]);
                float r0 = 1.0f / (1.0f + __expf(-ar0[i]));
                float z0 = 1.0f / (1.0f + __expf(-az0[i]));
                float n0 = tanhf(ain0[i] + r0 * ahn0[i]);
                float r1 = 1.0f / (1.0f + __expf(-ar1[i]));
                float z1 = 1.0f / (1.0f + __expf(-az1[i]));
                float n1 = tanhf(ain1[i] + r1 * ahn1[i]);
                nxt[m * 1024 + ca] = f2bf((1.0f - z0) * n0 + z0 * h0v);
                nxt[m * 1024 + cb] = f2bf((1.0f - z1) * n1 + z1 * h1v);
            }
        }
        __syncthreads();

        // ========= Phase B: hid = relu(h_new @ Wp^T + bp), paired tiles =========
        for (int pp = 0; pp < 4; ++pp) {
            int ta = w * 8 + pp * 2;
            int tb = ta + 1;
            int ca = ta * 16 + lm;
            int cb = tb * 16 + lm;
            float bb0 = bp[ca], bb1 = bp[cb];
            v8f acc0, acc1;
            for (int i = 0; i < 8; ++i) { acc0[i] = bb0; acc1[i] = bb1; }
            for (int kt = 0; kt < 32; ++kt) {
                prefetch_b(pkWp, (long)ta * 32 + kt + 8, lane);
                v16bf a = load_a(nxt, 1024, lm, hi, kt * 32);
                acc0 = wmma_bf16(a, load_b(pkWp, (long)ta * 32 + kt, lane), acc0);
                acc1 = wmma_bf16(a, load_b(pkWp, (long)tb * 32 + kt, lane), acc1);
            }
            for (int i = 0; i < 8; ++i) {
                int m = hi * 8 + i;
                hid[m * 1024 + ca] = f2bf(acc0[i] > 0.0f ? acc0[i] : 0.0f);
                hid[m * 1024 + cb] = f2bf(acc1[i] > 0.0f ? acc1[i] : 0.0f);
            }
        }
        __syncthreads();

        // ======= Phase C: per-joint heads, wave w owns joint j = rr*8 + w =======
        for (int rr = 0; rr < 2; ++rr) {
            int j = rr * 8 + w;
            if (j < 15) {   // wave-uniform branch: EXEC all-ones for WMMA
                unsigned short* uj = ub + w * 16 * 128;
                long baseW1 = (long)j * 8 * 32;
                // u_j = relu(hid @ W1[j] + b1[j])  [16 x 128], paired N-tiles
                for (int pp = 0; pp < 4; ++pp) {
                    int tna = pp * 2, tnb = tna + 1;
                    int ca = tna * 16 + lm, cb = tnb * 16 + lm;
                    float bb0 = b1[j * 128 + ca], bb1 = b1[j * 128 + cb];
                    v8f acc0, acc1;
                    for (int i = 0; i < 8; ++i) { acc0[i] = bb0; acc1[i] = bb1; }
                    for (int kt = 0; kt < 32; ++kt) {
                        prefetch_b(pkW1, baseW1 + (long)tna * 32 + kt + 8, lane);
                        v16bf a = load_a(hid, 1024, lm, hi, kt * 32);
                        acc0 = wmma_bf16(a, load_b(pkW1, baseW1 + (long)tna * 32 + kt, lane), acc0);
                        acc1 = wmma_bf16(a, load_b(pkW1, baseW1 + (long)tnb * 32 + kt, lane), acc1);
                    }
                    for (int i = 0; i < 8; ++i) {
                        int m = hi * 8 + i;
                        uj[m * 128 + ca] = f2bf(acc0[i] > 0.0f ? acc0[i] : 0.0f);
                        uj[m * 128 + cb] = f2bf(acc1[i] > 0.0f ? acc1[i] : 0.0f);
                    }
                }
                // delta_j = u_j @ W2[j] + b2[j]  [16 x 16], cols 0..8 valid
                v8f acc;
                float bb = (lm < 9) ? b2[j * 9 + lm] : 0.0f;
                for (int i = 0; i < 8; ++i) acc[i] = bb;
                long base2 = (long)j * 4;
                for (int kt = 0; kt < 4; ++kt)
                    acc = wmma_bf16(load_a(uj, 128, lm, hi, kt * 32),
                                    load_b(pkW2, base2 + kt, lane), acc);
                if (lm < 9) {
                    for (int i = 0; i < 8; ++i) {
                        int m  = hi * 8 + i;
                        int xi = j * 9 + lm;
                        float v = xf[m * 160 + xi] + acc[i];
                        xf[m * 160 + xi] = v;
                        xb[m * 160 + xi] = f2bf(v);
                        out[(((long)(b0 + m)) * 24 + f) * 135 + xi] = v;
                    }
                }
            }
        }
        __syncthreads();
    }
}

// ---------------------------------------------------------------------------
// Host launcher
// ---------------------------------------------------------------------------
extern "C" void kernel_launch(void* const* d_in, const int* in_sizes, int n_in,
                              void* d_out, int out_size, void* d_ws, size_t ws_size,
                              hipStream_t stream) {
    const float* poses = (const float*)d_in[0];
    const float* W_ih  = (const float*)d_in[1];
    const float* W_hh  = (const float*)d_in[2];
    const float* b_ih  = (const float*)d_in[3];
    const float* b_hh  = (const float*)d_in[4];
    const float* Wp    = (const float*)d_in[5];
    const float* bp    = (const float*)d_in[6];
    const float* W1    = (const float*)d_in[7];
    const float* b1    = (const float*)d_in[8];
    const float* W2    = (const float*)d_in[9];
    const float* b2    = (const float*)d_in[10];

    // Workspace carve (bf16 halfs): pre-swizzled B-fragment weights, ~12.8 MB.
    unsigned short* ws    = (unsigned short*)d_ws;
    unsigned short* pkWih = ws;                       // 192*5*512   = 491520
    unsigned short* pkWhh = pkWih + 491520;           // 192*32*512  = 3145728
    unsigned short* pkWp  = pkWhh + 3145728;          // 64*32*512   = 1048576
    unsigned short* pkW1  = pkWp  + 1048576;          // 15*8*32*512 = 1966080
    unsigned short* pkW2  = pkW1  + 1966080;          // 15*1*4*512  = 30720

    auto pack = [&](const float* src, unsigned short* dst, int NT, int KT,
                    int Ns, int Ks, long sn, long sk) {
        long total = (long)NT * KT * 512;
        int  thr   = 256;
        int  blk   = (int)((total + thr - 1) / thr);
        pack_b_kernel<<<blk, thr, 0, stream>>>(src, dst, NT, KT, Ns, Ks, sn, sk);
    };

    // B[k][n] = W[n][k] for x@W^T style matmuls (row-major weights).
    pack(W_ih, pkWih, 192, 5, 3072, 135, /*sn=*/135,  /*sk=*/1);
    pack(W_hh, pkWhh, 192, 32, 3072, 1024, /*sn=*/1024, /*sk=*/1);
    pack(Wp,   pkWp,   64, 32, 1024, 1024, /*sn=*/1024, /*sk=*/1);
    // W1[j]: B[k=h][n] = W1[j][h][n]  -> sn=1, sk=128
    for (int j = 0; j < 15; ++j)
        pack(W1 + (long)j * 1024 * 128, pkW1 + (long)j * 8 * 32 * 512,
             8, 32, 128, 1024, /*sn=*/1, /*sk=*/128);
    // W2[j]: B[k][d] = W2[j][k][d] padded to N=16 -> sn=1, sk=9
    for (int j = 0; j < 15; ++j)
        pack(W2 + (long)j * 128 * 9, pkW2 + (long)j * 4 * 512,
             1, 4, 9, 128, /*sn=*/1, /*sk=*/9);

    // Dynamic LDS: h x2 + hid + u (bf16) + x (f32 + bf16) = 146432 B < 320 KB.
    size_t shmem = (size_t)(16 * 1024 * 3 + 8 * 16 * 128) * sizeof(unsigned short)
                 + (size_t)16 * 160 * sizeof(float)
                 + (size_t)16 * 160 * sizeof(unsigned short);

    gru_rollout<<<128, 256, shmem, stream>>>(
        poses, b_ih, b_hh, bp, b1, b2,
        pkWih, pkWhh, pkWp, pkW1, pkW2, (float*)d_out);
}
